// Impaint_42451456753728
// MI455X (gfx1250) — matrix operand order repaired and checked by reading, not compiled, not run on hardware
//
#include <hip/hip_runtime.h>

typedef unsigned short u16;
typedef unsigned int   u32;

typedef __attribute__((ext_vector_type(16))) __bf16 v16bf;
typedef __attribute__((ext_vector_type(8)))  float  v8f;

#define NVERT 49152
#define NBATCH 16
#define NDEG 8

__device__ __forceinline__ u16 f2bfu(float f) {
  union { float f; u32 u; } v; v.f = f;
  u32 u = v.u;
  u32 r = u + 0x7FFFu + ((u >> 16) & 1u);   // round-to-nearest-even
  return (u16)(r >> 16);
}
__device__ __forceinline__ float bfu2f(u16 s) {
  union { u32 u; float f; } v; v.u = ((u32)s) << 16; return v.f;
}
__device__ __forceinline__ u32 pk2(float a, float b) {
  return (u32)f2bfu(a) | ((u32)f2bfu(b) << 16);
}
// acc[0..7] += w * unpack8(p)
__device__ __forceinline__ void fma8(uint4 p, float w, float* a) {
  a[0] += w * bfu2f((u16)p.x); a[1] += w * bfu2f((u16)(p.x >> 16));
  a[2] += w * bfu2f((u16)p.y); a[3] += w * bfu2f((u16)(p.y >> 16));
  a[4] += w * bfu2f((u16)p.z); a[5] += w * bfu2f((u16)(p.z >> 16));
  a[6] += w * bfu2f((u16)p.w); a[7] += w * bfu2f((u16)(p.w >> 16));
}

// ---------------------------------------------------------------------------
// Layer-1 helpers (Fin=1: too skinny for WMMA, pure VALU)
// ---------------------------------------------------------------------------

// h0[v*B+b] = x[b*V+v]   (f32)
__global__ void __launch_bounds__(256) k_transpose(const float* __restrict__ x,
                                                   float* __restrict__ h0) {
  int tid = blockIdx.x * blockDim.x + threadIdx.x;     // tid = b*V + v (coalesced read)
  int b = tid / NVERT, v = tid % NVERT;
  h0[(size_t)v * NBATCH + b] = x[tid];
}

// x1[v*B+b] = sum_j w[v*8+j] * h0[col[v*8+j]*B + b]   (f32, Fin=1)
__global__ void __launch_bounds__(256) k_spmm1(const float* __restrict__ h0,
                                               const int* __restrict__ cols,
                                               const float* __restrict__ wts,
                                               float* __restrict__ x1) {
  int tid = blockIdx.x * blockDim.x + threadIdx.x;     // tid = v*B + b
  int v = tid / NBATCH, b = tid % NBATCH;
  const int* cp = cols + v * NDEG;
  const float* wp = wts + v * NDEG;
  float acc = 0.0f;
#pragma unroll
  for (int j = 0; j < NDEG; ++j)
    acc += wp[j] * h0[(size_t)cp[j] * NBATCH + b];
  x1[tid] = acc;
}

// x2 = 2*L*x1 - x0 ; out_l1[v,b,o] = relu(x0*w1[0,o] + x1*w1[1,o] + x2*w1[2,o] + b1[o])
// Writes layer-2 Xcat slot k=0 (channels 0..15) and zeroes pad channels 48..63.
__global__ void __launch_bounds__(256) k_l1_finish(const float* __restrict__ h0,
                                                   const float* __restrict__ x1,
                                                   const int* __restrict__ cols,
                                                   const float* __restrict__ wts,
                                                   const float* __restrict__ w1,
                                                   const float* __restrict__ b1,
                                                   u16* __restrict__ xcat2) {
  int tid = blockIdx.x * blockDim.x + threadIdx.x;     // tid = v*B + b
  int v = tid / NBATCH;
  const int* cp = cols + v * NDEG;
  const float* wp = wts + v * NDEG;
  float s = 0.0f;
#pragma unroll
  for (int j = 0; j < NDEG; ++j)
    s += wp[j] * x1[(size_t)cp[j] * NBATCH + (tid % NBATCH)];
  float c0 = h0[tid];
  float c1 = x1[tid];
  float c2 = 2.0f * s - c0;
  u16* row = xcat2 + (size_t)tid * 64;
  u32 q[8];
#pragma unroll
  for (int o = 0; o < 16; o += 2) {
    float a = fmaxf(c0 * w1[o]     + c1 * w1[16 + o]     + c2 * w1[32 + o]     + b1[o],     0.0f);
    float d = fmaxf(c0 * w1[o + 1] + c1 * w1[16 + o + 1] + c2 * w1[32 + o + 1] + b1[o + 1], 0.0f);
    q[o >> 1] = pk2(a, d);
  }
  uint4 s0 = {q[0], q[1], q[2], q[3]};
  uint4 s1 = {q[4], q[5], q[6], q[7]};
  uint4 z  = {0u, 0u, 0u, 0u};
  *(uint4*)(row)      = s0;                  // channels 0..7
  *(uint4*)(row + 8)  = s1;                  // channels 8..15
  *(uint4*)(row + 48) = z;                   // K-pad 48..55
  *(uint4*)(row + 56) = z;                   // K-pad 56..63
}

// ---------------------------------------------------------------------------
// Generic SpMM on bf16 activations stored [V, B, ld] (ld = padded K-dim).
// y[v,b,:] = alpha * sum_j w_j * x[col_j, b, :]  (+ beta * z[v,b,:])
// One thread per (v, b, 8-channel group): every gather is a 16-byte
// global_load_b128, the store a 16-byte b128 — address math amortized 4x
// vs dword gathers; this is the dominant-traffic path.
// ---------------------------------------------------------------------------
template <int FIN>
__global__ void __launch_bounds__(256) k_spmm_bf16(const u16* __restrict__ xin,
                                                   const u16* __restrict__ zin,
                                                   u16* __restrict__ yout, int ld,
                                                   const int* __restrict__ cols,
                                                   const float* __restrict__ wts,
                                                   float alpha, float beta) {
  constexpr int F8 = FIN / 8;                          // 8-channel groups per (v,b)
  int tid = blockIdx.x * blockDim.x + threadIdx.x;
  int f8 = tid % F8;
  int b = (tid / F8) % NBATCH;
  int v = tid / (F8 * NBATCH);
  const int* cp = cols + v * NDEG;
  const float* wp = wts + v * NDEG;
  float a[8];
#pragma unroll
  for (int i = 0; i < 8; ++i) a[i] = 0.0f;
#pragma unroll
  for (int j = 0; j < NDEG; ++j) {
    int c = cp[j];
    float w = wp[j];
    uint4 p = *(const uint4*)(xin + ((size_t)c * NBATCH + b) * ld + f8 * 8);
    fma8(p, w, a);
  }
  size_t rowoff = ((size_t)v * NBATCH + b) * ld + f8 * 8;
  float y[8];
#pragma unroll
  for (int i = 0; i < 8; ++i) y[i] = alpha * a[i];
  if (zin) {
    uint4 zp = *(const uint4*)(zin + rowoff);
    float zf[8];
#pragma unroll
    for (int i = 0; i < 8; ++i) zf[i] = 0.0f;
    fma8(zp, 1.0f, zf);
#pragma unroll
    for (int i = 0; i < 8; ++i) y[i] += beta * zf[i];
  }
  uint4 o;
  o.x = pk2(y[0], y[1]); o.y = pk2(y[2], y[3]);
  o.z = pk2(y[4], y[5]); o.w = pk2(y[6], y[7]);
  *(uint4*)(yout + rowoff) = o;
}

// ---------------------------------------------------------------------------
// WMMA GEMM: C[M, N] = A[M, LDA](bf16) x W[Kd, N](f32) + bias, opt relu.
//   LDA = KSTEPS*32 (zero-padded K dim), N = NTILES*16.
// One wave computes one full 16-row x N strip: A fragment loaded once per
// k-step and reused across NTILES wmmas.
// W is staged once per block into LDS, pre-swizzled into the native 16-bit
// B 32x16 striping (lane l -> col l%16; halves h -> K = (l<16?0:8)+(h&7)+(h&8?16:0)), so the
// hot loop is: 2x global_load_b128 (A) + 2x ds_load_b128 (B) + v_wmma.
// A fragment striping (16-bit A 16x32): lane l, row = l%16,
//   halves 0..7 -> K = kb+0..7, halves 8..15 -> K = 16+kb+0..7, kb=(l<16)?0:8.
// C/D: lane l col = l%16, vgpr r -> row = r + (l<16?0:8).
// ---------------------------------------------------------------------------
template <int KSTEPS, int NTILES>
__global__ void __launch_bounds__(256) k_gemm_wmma(const u16* __restrict__ A,
                                                   const float* __restrict__ W,
                                                   const float* __restrict__ bias,
                                                   int Kd,
                                                   u16* __restrict__ out, int ldc,
                                                   int relu, int totalWaves) {
  constexpr int LDA = KSTEPS * 32;
  constexpr int N = NTILES * 16;
  constexpr int BSTAGE = KSTEPS * NTILES * 512;        // u16 elements (32B per lane-frag)
  __shared__ u16 bstage[BSTAGE];

  // One-time cooperative stage: W (f32, Kd x N) -> bf16 B fragments in LDS.
  // element e -> (s, t, lane, h); K-pad rows (k >= Kd) become zero here.
  for (int e = threadIdx.x; e < BSTAGE; e += 256) {
    int h = e & 15;
    int lane = (e >> 4) & 31;
    int t = (e >> 9) % NTILES;
    int s = e / (512 * NTILES);
    int k = s * 32 + ((lane & 16) ? 8 : 0) + (h & 7) + ((h & 8) ? 16 : 0);
    int n = t * 16 + (lane & 15);
    float w = (k < Kd) ? W[(size_t)k * N + n] : 0.0f;
    bstage[e] = f2bfu(w);
  }
  __syncthreads();

  int wid = blockIdx.x * 8 + (threadIdx.x >> 5);       // wave id == M-tile id
  if (wid >= totalWaves) return;                       // wave-uniform (grids exact)
  int lane = threadIdx.x & 31;
  int hi = lane >> 4;                                  // half-wave select
  int ln = lane & 15;
  int akb = hi ? 8 : 0;
  const u16* arow = A + (size_t)(wid * 16 + ln) * LDA + akb;

  v8f acc[NTILES];
  v8f zero = {};
#pragma unroll
  for (int t = 0; t < NTILES; ++t) acc[t] = zero;

#pragma unroll
  for (int s = 0; s < KSTEPS; ++s) {
    union { uint4 u[2]; v16bf v; } af;
    const u16* ap = arow + s * 32;
    af.u[0] = *(const uint4*)(ap);                     // halves 0..7
    af.u[1] = *(const uint4*)(ap + 16);                // halves 8..15
#pragma unroll
    for (int t = 0; t < NTILES; ++t) {
      union { uint4 u[2]; v16bf v; } bfr;
      const u16* bp = &bstage[((s * NTILES + t) * 32 + lane) * 16];
      bfr.u[0] = *(const uint4*)(bp);
      bfr.u[1] = *(const uint4*)(bp + 8);
      acc[t] = __builtin_amdgcn_wmma_f32_16x16x32_bf16(false, af.v, false, bfr.v,
                                                       (short)0, acc[t], false, false);
    }
  }

#pragma unroll
  for (int t = 0; t < NTILES; ++t) {
    int n = t * 16 + ln;
    float bv = bias[n];
    size_t obase = (size_t)(wid * 16 + hi * 8) * ldc + n;
#pragma unroll
    for (int r = 0; r < 8; ++r) {
      float v = acc[t][r] + bv;
      if (relu) v = fmaxf(v, 0.0f);
      out[obase + (size_t)r * ldc] = f2bfu(v);
    }
  }
}

// ---------------------------------------------------------------------------
// Layer 4: K=1, Fin=16, Fout=1 pointwise + transpose back to [B, V].
// ---------------------------------------------------------------------------
__global__ void __launch_bounds__(256) k_layer4(const u16* __restrict__ H3,
                                                const float* __restrict__ w4,
                                                const float* __restrict__ b4,
                                                float* __restrict__ out) {
  int tid = blockIdx.x * blockDim.x + threadIdx.x;   // tid = b*V + v (coalesced write)
  int b = tid / NVERT, v = tid % NVERT;
  const u16* h = H3 + ((size_t)v * NBATCH + b) * 16;
  uint4 p0 = *(const uint4*)(h);
  uint4 p1 = *(const uint4*)(h + 8);
  float a[8];
#pragma unroll
  for (int i = 0; i < 8; ++i) a[i] = 0.0f;
  fma8(p0, 1.0f, a);                                  // a[i]   = h[i],   i<8 (w=1 unpack)
  float acc = b4[0];
#pragma unroll
  for (int i = 0; i < 8; ++i) acc += a[i] * w4[i];
#pragma unroll
  for (int i = 0; i < 8; ++i) a[i] = 0.0f;
  fma8(p1, 1.0f, a);                                  // a[i] = h[8+i]
#pragma unroll
  for (int i = 0; i < 8; ++i) acc += a[i] * w4[8 + i];
  out[tid] = acc;
}

// ---------------------------------------------------------------------------
// Launcher
// ---------------------------------------------------------------------------
extern "C" void kernel_launch(void* const* d_in, const int* in_sizes, int n_in,
                              void* d_out, int out_size, void* d_ws, size_t ws_size,
                              hipStream_t stream) {
  (void)in_sizes; (void)n_in; (void)out_size; (void)ws_size;
  const float* x    = (const float*)d_in[0];
  // d_in[1] = edge_row: implied by fixed degree-8 row-sorted layout, unused
  const int*   ecol = (const int*)d_in[2];
  const float* ew   = (const float*)d_in[3];
  const float* w1   = (const float*)d_in[4];
  const float* b1   = (const float*)d_in[5];
  const float* w2   = (const float*)d_in[6];
  const float* b2   = (const float*)d_in[7];
  const float* w3   = (const float*)d_in[8];
  const float* b3   = (const float*)d_in[9];
  const float* w4   = (const float*)d_in[10];
  const float* b4   = (const float*)d_in[11];

  const size_t nvb = (size_t)NVERT * NBATCH;           // 786432 rows (M dim)
  char* ws = (char*)d_ws;

  // Workspace layout (with liveness-based aliasing):
  //   [0,            100663296) Xcat2: [V,B,64]  bf16  (layer-2 Cheb concat, Kd=48 pad 64)
  //   [100663296,    402653184) Xcat3: [V,B,192] bf16  (layer-3 Cheb concat, Kd=192)
  //   h0/x1 (f32, layer 1) alias the Xcat3 region (dead before Xcat3 is written).
  //   H3 ([V,B,16] bf16, layer-3 output) aliases the Xcat2 region (dead by then).
  u16*   Xcat2 = (u16*)ws;
  char*  p3    = ws + nvb * 64 * 2;
  u16*   Xcat3 = (u16*)p3;
  float* h0    = (float*)p3;
  float* x1l1  = (float*)(p3 + nvb * 4);
  u16*   H3    = (u16*)ws;

  const int T = 256;
  const int gVB = (int)(nvb / T);                      // 3072

  // ---- layer 1 (Fin=1, Fout=16, K=3) ----
  k_transpose<<<gVB, T, 0, stream>>>(x, h0);
  k_spmm1<<<gVB, T, 0, stream>>>(h0, ecol, ew, x1l1);
  k_l1_finish<<<gVB, T, 0, stream>>>(h0, x1l1, ecol, ew, w1, b1, Xcat2);

  // ---- layer 2 (Fin=16, Fout=64, K=3): Kd=48, Kdp=64, N=64 ----
  k_spmm_bf16<16><<<(int)(nvb * 2 / T), T, 0, stream>>>(
      Xcat2 + 0, nullptr, Xcat2 + 16, 64, ecol, ew, 1.0f, 0.0f);
  k_spmm_bf16<16><<<(int)(nvb * 2 / T), T, 0, stream>>>(
      Xcat2 + 16, Xcat2 + 0, Xcat2 + 32, 64, ecol, ew, 2.0f, -1.0f);
  {
    int waves = (int)(nvb / 16);                       // 49152 M-tiles, 1 wave each
    k_gemm_wmma<2, 4><<<waves / 8, T, 0, stream>>>(Xcat2, w2, b2, 48,
                                                   Xcat3, 192, 1, waves);
  }

  // ---- layer 3 (Fin=64, Fout=16, K=3): Kd=Kdp=192, N=16 ----
  k_spmm_bf16<64><<<(int)(nvb * 8 / T), T, 0, stream>>>(
      Xcat3 + 0, nullptr, Xcat3 + 64, 192, ecol, ew, 1.0f, 0.0f);
  k_spmm_bf16<64><<<(int)(nvb * 8 / T), T, 0, stream>>>(
      Xcat3 + 64, Xcat3 + 0, Xcat3 + 128, 192, ecol, ew, 2.0f, -1.0f);
  {
    int waves = (int)(nvb / 16);                       // 49152 M-tiles
    k_gemm_wmma<6, 1><<<waves / 8, T, 0, stream>>>(Xcat3, w3, b3, 192,
                                                   H3, 16, 1, waves);
  }

  // ---- layer 4 (K=1, Fin=16, Fout=1) + transpose to [B, V] ----
  k_layer4<<<gVB, T, 0, stream>>>(H3, w4, b4, (float*)d_out);
}
